// Memory_72756745994889
// MI455X (gfx1250) — matrix operand order repaired and checked by reading, compiled, not run on hardware
//
#include <hip/hip_runtime.h>
#include <hip/hip_bf16.h>

typedef __attribute__((ext_vector_type(16))) __bf16 v16bf;
typedef __attribute__((ext_vector_type(8)))  __bf16 v8bf;
typedef __attribute__((ext_vector_type(8)))  float  v8f;

constexpr int B_  = 8;
constexpr int C_  = 512;
constexpr int HW_ = 64 * 64;   // 4096
constexpr int N_  = HW_;       // tokens per batch
constexpr int D_  = 512;       // feature dim
constexpr int M_  = 512;       // memory slots

__device__ __forceinline__ unsigned short f2bf(float f) {
  unsigned int u = __float_as_uint(f);
  u += 0x7FFFu + ((u >> 16) & 1u);           // round-to-nearest-even
  return (unsigned short)(u >> 16);
}

// ---------------------------------------------------------------- keys prep
__global__ void k_prep_keys(const float* __restrict__ keys,
                            unsigned short* __restrict__ kb,
                            unsigned short* __restrict__ kt) {
  int idx = blockIdx.x * 256 + threadIdx.x;          // M_*D_
  if (idx >= M_ * D_) return;
  int m = idx / D_, d = idx - m * D_;
  unsigned short v = f2bf(keys[idx]);
  kb[idx] = v;                                       // [M][D] bf16
  kt[d * M_ + m] = v;                                // [D][M] bf16 (transposed)
}

// ---------------------------------------------------------------- 1/||q||
__global__ void k_rnorm(const float* __restrict__ query, float* __restrict__ rnorm) {
  int idx = blockIdx.x * 256 + threadIdx.x;          // B_*N_
  if (idx >= B_ * N_) return;
  int b = idx / N_, n = idx - b * N_;
  const float* p = query + (size_t)b * C_ * HW_ + n;
  float s = 0.f;
  #pragma unroll 8
  for (int c = 0; c < C_; ++c) { float v = p[(size_t)c * HW_]; s += v * v; }
  rnorm[idx] = 1.f / fmaxf(sqrtf(s), 1e-12f);
}

// ------------------------------------------- qf[b,n,d] = q[b,d,n]*rnorm (LDS transpose)
__global__ void k_build_qf(const float* __restrict__ query, const float* __restrict__ rnorm,
                           float* __restrict__ qf, unsigned short* __restrict__ qf_bf) {
  __shared__ float tile[32][33];
  int b  = blockIdx.z;
  int n0 = blockIdx.x * 32, c0 = blockIdx.y * 32;
  int tx = threadIdx.x & 31, ty = threadIdx.x >> 5;  // 256 threads = 32x8
  const float* src = query + (size_t)b * C_ * HW_;
  #pragma unroll
  for (int j = 0; j < 4; ++j)
    tile[ty + 8 * j][tx] = src[(size_t)(c0 + ty + 8 * j) * HW_ + n0 + tx];
  __syncthreads();
  #pragma unroll
  for (int j = 0; j < 4; ++j) {
    int n = n0 + ty + 8 * j;
    float v = tile[tx][ty + 8 * j] * rnorm[b * N_ + n];
    size_t o = ((size_t)b * N_ + n) * D_ + c0 + tx;
    qf[o] = v;
    qf_bf[o] = f2bf(v);
  }
}

// ------------------------------ updated_query first half: channel layout == input layout
__global__ void k_out_qnorm(const float* __restrict__ query, const float* __restrict__ rnorm,
                            float* __restrict__ out0) {
  size_t idx = (size_t)blockIdx.x * 256 + threadIdx.x;
  if (idx >= (size_t)B_ * C_ * HW_) return;
  size_t b = idx / ((size_t)C_ * HW_);
  size_t rem = idx - b * (size_t)C_ * HW_;
  size_t c = rem / HW_;
  size_t n = rem - c * HW_;
  out0[(b * 2 * C_ + c) * HW_ + n] = query[idx] * rnorm[b * N_ + n];
}

// ---------------------------------------------------------------- GEMM1: score = qf·keysᵀ
__global__ void k_gemm_score(const __bf16* __restrict__ qf_bf, const __bf16* __restrict__ keys_bf,
                             float* __restrict__ score) {
  int wave = threadIdx.x >> 5, lane = threadIdx.x & 31;
  int b  = blockIdx.z;
  int n0 = blockIdx.x * 64 + (wave & 3) * 16;
  int m0 = blockIdx.y * 32 + (wave >> 2) * 16;
  const __bf16* A  = qf_bf + (size_t)b * N_ * D_ +
                     (size_t)(n0 + (lane & 15)) * D_ + ((lane >> 4) * 16);
  const __bf16* Bp = keys_bf + (size_t)(m0 + (lane & 15)) * D_ + ((lane >> 4) * 16);
  v8f acc = {};
  #pragma unroll 4
  for (int kk = 0; kk < D_; kk += 32) {
    v8bf alo = *(const v8bf*)(A + kk);
    v8bf ahi = *(const v8bf*)(A + kk + 8);
    v16bf a = __builtin_shufflevector(alo, ahi, 0,1,2,3,4,5,6,7,8,9,10,11,12,13,14,15);
    v16bf bb = *(const v16bf*)(Bp + kk);
    acc = __builtin_amdgcn_wmma_f32_16x16x32_bf16(false, a, false, bb, (short)0, acc,
                                                  false, false);
  }
  float* Cb = score + (size_t)b * N_ * M_;
  int cn = n0 + 8 * (lane >> 4), cm = m0 + (lane & 15);
  #pragma unroll
  for (int r = 0; r < 8; ++r) Cb[(size_t)(cn + r) * M_ + cm] = acc[r];
}

// ---------------------------------------------------------------- colmax over n
__global__ void k_colmax(const float* __restrict__ score, float* __restrict__ colmax) {
  int m = blockIdx.x * 256 + threadIdx.x;
  int b = blockIdx.y;
  const float* S = score + (size_t)b * N_ * M_ + m;
  float v = -3.4e38f;
  for (int n = 0; n < N_; ++n) v = fmaxf(v, S[(size_t)n * M_]);
  colmax[b * M_ + m] = v;
}

// ---------------------------------------------------------------- top-2 merge helper
__device__ __forceinline__ void top2_merge(float& m1, int& x1, float& m2, int& x2,
                                           float n1, int y1, float n2, int y2) {
  if (n1 > m1 || (n1 == m1 && y1 < x1)) {
    float c = m1; int cx = x1;
    m1 = n1; x1 = y1;
    if (c > n2 || (c == n2 && cx < y2)) { m2 = c;  x2 = cx; }
    else                               { m2 = n2; x2 = y2; }
  } else if (n1 > m2 || (n1 == m2 && y1 < x2)) {
    m2 = n1; x2 = y1;
  }
}

// ------------------------- row softmax (over M) + top-2 indices + raw rowmax, bf16 out
__global__ void k_row_softmax(const float* __restrict__ score,
                              unsigned short* __restrict__ smem_bf,
                              int* __restrict__ gi1, int* __restrict__ gi2,
                              float* __restrict__ sassign) {
  __shared__ float sv1[128], sv2[128], ssum[128];
  __shared__ int   si1[128], si2[128];
  int n = blockIdx.x, b = blockIdx.y, t = threadIdx.x;
  const float* S = score + ((size_t)b * N_ + n) * M_;
  float s[4];
  float m1 = -3.4e38f, m2 = -3.4e38f; int x1 = 0x7fffffff, x2 = 0x7fffffff;
  #pragma unroll
  for (int j = 0; j < 4; ++j) {
    int mm = t + 128 * j;
    s[j] = S[mm];
    top2_merge(m1, x1, m2, x2, s[j], mm, -3.4e38f, 0x7fffffff);
  }
  sv1[t] = m1; si1[t] = x1; sv2[t] = m2; si2[t] = x2;
  __syncthreads();
  for (int st = 64; st > 0; st >>= 1) {
    if (t < st) {
      float a1 = sv1[t], a2 = sv2[t]; int i1 = si1[t], i2 = si2[t];
      top2_merge(a1, i1, a2, i2, sv1[t + st], si1[t + st], sv2[t + st], si2[t + st]);
      sv1[t] = a1; si1[t] = i1; sv2[t] = a2; si2[t] = i2;
    }
    __syncthreads();
  }
  float rowmax = sv1[0];
  int top1 = si1[0], top2i = si2[0];
  float e[4], lsum = 0.f;
  #pragma unroll
  for (int j = 0; j < 4; ++j) { e[j] = expf(s[j] - rowmax); lsum += e[j]; }
  ssum[t] = lsum; __syncthreads();
  for (int st = 64; st > 0; st >>= 1) { if (t < st) ssum[t] += ssum[t + st]; __syncthreads(); }
  float inv = 1.f / ssum[0];
  unsigned short* O = smem_bf + ((size_t)b * N_ + n) * M_;
  #pragma unroll
  for (int j = 0; j < 4; ++j) O[t + 128 * j] = f2bf(e[j] * inv);
  if (t == 0) {
    gi1[b * N_ + n] = top1;
    gi2[b * N_ + n] = top2i;
    sassign[b * N_ + n] = rowmax;   // score[b,n,argmax] == rowmax
  }
}

// --------------------------- GEMM2: concat = smem·keys, LDS-transposed coalesced store
__global__ void k_gemm_read(const __bf16* __restrict__ smem_bf,
                            const __bf16* __restrict__ keysT_bf,
                            float* __restrict__ out0) {
  __shared__ float tile[32][65];
  int wave = threadIdx.x >> 5, lane = threadIdx.x & 31;
  int b   = blockIdx.z;
  int n0l = (wave & 3) * 16, d0l = (wave >> 2) * 16;
  int n0  = blockIdx.x * 64, d0 = blockIdx.y * 32;
  const __bf16* A  = smem_bf + (size_t)b * N_ * M_ +
                     (size_t)(n0 + n0l + (lane & 15)) * M_ + ((lane >> 4) * 16);
  const __bf16* Bp = keysT_bf + (size_t)(d0 + d0l + (lane & 15)) * M_ + ((lane >> 4) * 16);
  v8f acc = {};
  #pragma unroll 4
  for (int kk = 0; kk < M_; kk += 32) {
    v8bf alo = *(const v8bf*)(A + kk);
    v8bf ahi = *(const v8bf*)(A + kk + 8);
    v16bf a = __builtin_shufflevector(alo, ahi, 0,1,2,3,4,5,6,7,8,9,10,11,12,13,14,15);
    v16bf bb = *(const v16bf*)(Bp + kk);
    acc = __builtin_amdgcn_wmma_f32_16x16x32_bf16(false, a, false, bb, (short)0, acc,
                                                  false, false);
  }
  int cn = n0l + 8 * (lane >> 4), cd = d0l + (lane & 15);
  #pragma unroll
  for (int r = 0; r < 8; ++r) tile[cd][cn + r] = acc[r];
  __syncthreads();
  float* Ob = out0 + ((size_t)b * 2 * C_ + C_ + d0) * HW_ + n0;
  int nl = threadIdx.x & 63, ds = threadIdx.x >> 6;
  for (int dl = ds; dl < 32; dl += 4)
    Ob[(size_t)dl * HW_ + nl] = tile[dl][nl];
}

// ---------------------------------------------------------------- losses
__global__ void k_loss(const float* __restrict__ qf, const float* __restrict__ keys,
                       const int* __restrict__ gi1, const int* __restrict__ gi2,
                       float* __restrict__ o_lsep, float* __restrict__ o_lcom,
                       float* __restrict__ o_idx) {
  __shared__ float r1[128], r2[128];
  int n = blockIdx.x, b = blockIdx.y, t = threadIdx.x;
  int idx = b * N_ + n;
  int i1 = gi1[idx], i2 = gi2[idx];
  const float* q = qf + (size_t)idx * D_;
  const float* p = keys + (size_t)i1 * D_;
  const float* g = keys + (size_t)i2 * D_;
  float* lc = o_lcom + (size_t)idx * D_;
  float sp = 0.f, sn = 0.f;
  #pragma unroll
  for (int j = 0; j < 4; ++j) {
    int d = t + 128 * j;
    float qd = q[d], dp = qd - p[d], dn = qd - g[d];
    lc[d] = dp * dp;
    float a = dp + 1e-6f, c = dn + 1e-6f;
    sp += a * a; sn += c * c;
  }
  r1[t] = sp; r2[t] = sn; __syncthreads();
  for (int st = 64; st > 0; st >>= 1) {
    if (t < st) { r1[t] += r1[t + st]; r2[t] += r2[t + st]; }
    __syncthreads();
  }
  if (t == 0) {
    o_lsep[idx] = fmaxf(sqrtf(r1[0]) - sqrtf(r2[0]) + 1.0f, 0.f);
    o_idx[idx] = (float)i1;
  }
}

// ------------------------- one-hot weighted scatter: qu[b,assign,:] += w*qf[b,n,:]
__global__ void k_scatter_update(const float* __restrict__ qf, const int* __restrict__ gi1,
                                 const float* __restrict__ sassign,
                                 const float* __restrict__ colmax,
                                 float* __restrict__ qu) {
  int n = blockIdx.x, b = blockIdx.y, t = threadIdx.x;
  int idx = b * N_ + n;
  int m = gi1[idx];
  float w = expf(sassign[idx] - colmax[b * M_ + m]);  // score_query/max_sq simplification
  const float* q = qf + (size_t)idx * D_;
  float* dst = qu + ((size_t)b * M_ + m) * D_;
  #pragma unroll
  for (int j = 0; j < 4; ++j) {
    int d = t + 128 * j;
    atomicAdd(&dst[d], w * q[d]);
  }
}

// ------------------------- per-row 8-step accumulate+renormalize scan
__global__ void k_update_memory(const float* __restrict__ keys, const float* __restrict__ qu,
                                float* __restrict__ o_mem) {
  __shared__ float red[128];
  int m = blockIdx.x, t = threadIdx.x;
  float v[4];
  #pragma unroll
  for (int j = 0; j < 4; ++j) v[j] = keys[(size_t)m * D_ + t + 128 * j];
  for (int b = 0; b < B_; ++b) {
    const float* qb = qu + ((size_t)b * M_ + m) * D_;
    float s = 0.f;
    #pragma unroll
    for (int j = 0; j < 4; ++j) { v[j] += qb[t + 128 * j]; s += v[j] * v[j]; }
    red[t] = s; __syncthreads();
    for (int st = 64; st > 0; st >>= 1) { if (t < st) red[t] += red[t + st]; __syncthreads(); }
    float inv = 1.f / fmaxf(sqrtf(red[0]), 1e-12f);
    __syncthreads();
    #pragma unroll
    for (int j = 0; j < 4; ++j) v[j] *= inv;
  }
  #pragma unroll
  for (int j = 0; j < 4; ++j) o_mem[(size_t)m * D_ + t + 128 * j] = v[j];
}

// ---------------------------------------------------------------- launch
extern "C" void kernel_launch(void* const* d_in, const int* in_sizes, int n_in,
                              void* d_out, int out_size, void* d_ws, size_t ws_size,
                              hipStream_t stream) {
  const float* query = (const float*)d_in[0];   // [8,512,64,64]
  const float* keys  = (const float*)d_in[1];   // [512,512]

  // workspace carve-up
  char* w = (char*)d_ws;
  float*          qf      = (float*)w;                 w += (size_t)B_ * N_ * D_ * 4;  // 64 MB
  float*          score   = (float*)w;                 w += (size_t)B_ * N_ * M_ * 4;  // 64 MB
  unsigned short* qf_bf   = (unsigned short*)w;        w += (size_t)B_ * N_ * D_ * 2;  // 32 MB
  unsigned short* smem_bf = (unsigned short*)w;        w += (size_t)B_ * N_ * M_ * 2;  // 32 MB
  unsigned short* keys_bf = (unsigned short*)w;        w += (size_t)M_ * D_ * 2;
  unsigned short* keysT_bf= (unsigned short*)w;        w += (size_t)D_ * M_ * 2;
  float*          qu      = (float*)w;                 w += (size_t)B_ * M_ * D_ * 4;  // 8 MB
  float*          rnorm   = (float*)w;                 w += (size_t)B_ * N_ * 4;
  float*          colmax  = (float*)w;                 w += (size_t)B_ * M_ * 4;
  int*            gi1     = (int*)w;                   w += (size_t)B_ * N_ * 4;
  int*            gi2     = (int*)w;                   w += (size_t)B_ * N_ * 4;
  float*          sassign = (float*)w;                 w += (size_t)B_ * N_ * 4;

  // outputs (concatenated flat, fp32)
  float* o0 = (float*)d_out;                 // updated_query [8,1024,64,64]
  float* o1 = o0 + (size_t)B_ * 2 * C_ * HW_;// updated_memory [512,512]
  float* o2 = o1 + (size_t)M_ * D_;          // loss_separate [8,4096]
  float* o3 = o2 + (size_t)B_ * N_;          // loss_compact [8,4096,512]
  float* o4 = o3 + (size_t)B_ * N_ * D_;     // closest indices (as float) [8,4096]

  k_prep_keys<<<(M_ * D_ + 255) / 256, 256, 0, stream>>>(keys, keys_bf, keysT_bf);
  k_rnorm<<<(B_ * N_ + 255) / 256, 256, 0, stream>>>(query, rnorm);
  k_build_qf<<<dim3(N_ / 32, C_ / 32, B_), 256, 0, stream>>>(query, rnorm, qf, qf_bf);
  k_out_qnorm<<<(int)(((size_t)B_ * C_ * HW_ + 255) / 256), 256, 0, stream>>>(query, rnorm, o0);
  k_gemm_score<<<dim3(N_ / 64, M_ / 32, B_), 256, 0, stream>>>(
      (const __bf16*)qf_bf, (const __bf16*)keys_bf, score);
  k_colmax<<<dim3(M_ / 256, B_), 256, 0, stream>>>(score, colmax);
  k_row_softmax<<<dim3(N_, B_), 128, 0, stream>>>(score, smem_bf, gi1, gi2, sassign);
  k_gemm_read<<<dim3(N_ / 64, D_ / 32, B_), 256, 0, stream>>>(
      (const __bf16*)smem_bf, (const __bf16*)keysT_bf, o0);
  k_loss<<<dim3(N_, B_), 128, 0, stream>>>(qf, keys, gi1, gi2, o2, o3, o4);
  hipMemsetAsync(qu, 0, (size_t)B_ * M_ * D_ * 4, stream);
  k_scatter_update<<<dim3(N_, B_), 128, 0, stream>>>(qf, gi1, sassign, colmax, qu);
  k_update_memory<<<M_, 128, 0, stream>>>(keys, qu, o1);
}
